// HMP_SchNetModel_77017353552165
// MI455X (gfx1250) — compile-verified, matrix operand order checked by reading, not run on hardware
//
#include <hip/hip_runtime.h>
#include <math.h>

#define Nn 2048
#define Ee 65536
#define Bb 8
#define Hh 128
#define Gg 50
#define Ff 64
#define Ss 16
#define MHh 32
#define Kk 512
#define CUTOFF_F 10.0f
#define PI_F 3.14159265358979f
#define GSTEP (CUTOFF_F / (float)(Gg - 1))
#define GCOEFF (-0.5f / (GSTEP * GSTEP))
#define LAMBDA_A 0.002f

typedef __attribute__((ext_vector_type(16))) __bf16    bf16x16;
typedef __attribute__((ext_vector_type(8)))  __bf16    bf16x8;
typedef __attribute__((ext_vector_type(8)))  float     floatx8;
typedef __attribute__((ext_vector_type(4)))  unsigned  uint4v;
typedef __attribute__((ext_vector_type(8)))  int       int8v;
typedef __attribute__((ext_vector_type(4)))  int       int4v;

// branchless softplus - ln2 on hardware transcendentals
static __device__ __forceinline__ float sspf(float x) {
  return fmaxf(x, 0.f) + __logf(1.f + __expf(-fabsf(x))) - 0.69314718056f;
}
static __device__ __forceinline__ float ccutf(float d) {
  return 0.5f * (__cosf(d * PI_F / CUTOFF_F) + 1.0f);
}
static __device__ __forceinline__ float sigmf(float x) {
  return 1.f / (1.f + __expf(-x));
}
// k index held by element j of a 16-bit WMMA A/B fragment (ISA 7.12.2)
static __device__ __forceinline__ int kmapf(int j, int half) {
  return (j < 8) ? (j + 8 * half) : (j + 8 + 8 * half);
}
// pack 16 floats from two contiguous float4-pairs into a bf16 fragment
static __device__ __forceinline__ bf16x16 pack_frag(float4 l0, float4 l1, float4 h0, float4 h1) {
  bf16x16 a;
  a[0] = (__bf16)l0.x; a[1] = (__bf16)l0.y; a[2]  = (__bf16)l0.z; a[3]  = (__bf16)l0.w;
  a[4] = (__bf16)l1.x; a[5] = (__bf16)l1.y; a[6]  = (__bf16)l1.z; a[7]  = (__bf16)l1.w;
  a[8] = (__bf16)h0.x; a[9] = (__bf16)h0.y; a[10] = (__bf16)h0.z; a[11] = (__bf16)h0.w;
  a[12] = (__bf16)h1.x; a[13] = (__bf16)h1.y; a[14] = (__bf16)h1.z; a[15] = (__bf16)h1.w;
  return a;
}

// ---- Tensor Data Mover: 1-D DMA of nElem f32 from global into LDS ----
// D# per CDNA5 ISA 8.3/8.4: group0 = {count=1, lds_addr, global_addr, type=2},
// group1 = {data_size=4B, tensor_dim0 = tile_dim0 = nElem, stride = nElem}.
// This toolchain exposes the 6-arg builtin (extra int32x8 group before cpol).
static __device__ __forceinline__ void tdm_load_f32(const float* gsrc, unsigned ldsAddr, int nElem) {
  unsigned long long ga = (unsigned long long)(uintptr_t)gsrc;
  uint4v g0;
  g0[0] = 1u;                                             // count=1 (valid), user mode
  g0[1] = ldsAddr;                                        // lds_addr [63:32]
  g0[2] = (unsigned)ga;                                   // global_addr low
  g0[3] = (unsigned)((ga >> 32) & 0x01FFFFFFu) | (2u << 30);  // global_addr[56:32] | type=2
  int8v g1;
  g1[0] = (2 << 16);                                      // data_size = 4B
  g1[1] = (int)((unsigned)(nElem & 0xFFFF) << 16);        // tensor_dim0[15:0] @ bits 63:48
  g1[2] = (int)((unsigned)nElem >> 16);                   // tensor_dim0[31:16] @ bits 79:64
  g1[3] = (int)((unsigned)(nElem & 0xFFFF) << 16);        // tile_dim0 @ bits 127:112
  g1[4] = 0;                                              // tile_dim1/2 = 0 (1-D tile)
  g1[5] = nElem;                                          // tensor_dim0_stride low
  g1[6] = 0;
  g1[7] = 0;
  int4v gz4 = {0, 0, 0, 0};                               // groups 2/3 unused (<=2-D)
  int8v gz8 = {0, 0, 0, 0, 0, 0, 0, 0};
  __builtin_amdgcn_tensor_load_to_lds(g0, g1, gz4, gz4, gz8, 0);
}
static __device__ __forceinline__ unsigned lds_off(const void* p) {
  return (unsigned)(uintptr_t)p;                          // low 32 bits = LDS byte offset
}

// ---------------- generic bf16-WMMA GEMM with fused epilogue ----------------
// C[M,Nb] = act(alpha * A[M,Ka] @ B + bias + resid); ACT: 0=none 1=ssp
template <int TRANSB, int ACT>
__global__ void gemm_ep_t(const float* __restrict__ A, int lda,
                          const float* __restrict__ Bm, int ldb,
                          const float* __restrict__ bias,
                          const float* __restrict__ resid,
                          float* __restrict__ C, int ldc,
                          int M, int Ka, int Nb, float alpha)
{
  int wave = blockIdx.x * (blockDim.x >> 5) + (threadIdx.x >> 5);
  int ntiles = Nb >> 4;
  int tot = (M >> 4) * ntiles;
  if (wave >= tot) return;
  int mt = wave / ntiles, nt = wave - mt * ntiles;
  int lane = threadIdx.x & 31, half = lane >> 4, r = lane & 15;
  int arow = (mt << 4) + r;
  int bcol = (nt << 4) + r;
  const float* ap = A + (long long)arow * lda;
  const float* bp = TRANSB ? (Bm + (long long)bcol * ldb) : (Bm + bcol);
  floatx8 c = {};
  int kfull = Ka & ~31;
  for (int k0 = 0; k0 < kfull; k0 += 32) {
    if (k0 + 32 < kfull) __builtin_prefetch(ap + k0 + 32, 0, 1);
    int kb = k0 + 8 * half;
    float4 al0 = *(const float4*)(ap + kb);
    float4 al1 = *(const float4*)(ap + kb + 4);
    float4 ah0 = *(const float4*)(ap + kb + 16);
    float4 ah1 = *(const float4*)(ap + kb + 20);
    bf16x16 a = pack_frag(al0, al1, ah0, ah1);
    bf16x16 b;
    if (TRANSB) {
      float4 bl0 = *(const float4*)(bp + kb);
      float4 bl1 = *(const float4*)(bp + kb + 4);
      float4 bh0 = *(const float4*)(bp + kb + 16);
      float4 bh1 = *(const float4*)(bp + kb + 20);
      b = pack_frag(bl0, bl1, bh0, bh1);
    } else {
#pragma unroll
      for (int j = 0; j < 16; ++j) {
        int k = k0 + kmapf(j, half);
        b[j] = (__bf16)bp[(long long)k * ldb];
      }
    }
    c = __builtin_amdgcn_wmma_f32_16x16x32_bf16(false, a, false, b, (short)0, c, false, false);
  }
  if (kfull < Ka) {            // remainder tile with bounds guards (Ka % 32 != 0)
    bf16x16 a, b;
#pragma unroll
    for (int j = 0; j < 16; ++j) {
      int k = kfull + kmapf(j, half);
      a[j] = (k < Ka) ? (__bf16)ap[k] : (__bf16)0.f;
      b[j] = (k < Ka) ? (TRANSB ? (__bf16)bp[k] : (__bf16)bp[(long long)k * ldb])
                      : (__bf16)0.f;
    }
    c = __builtin_amdgcn_wmma_f32_16x16x32_bf16(false, a, false, b, (short)0, c, false, false);
  }
#pragma unroll
  for (int v = 0; v < 8; ++v) {
    int row = (mt << 4) + v + 8 * half;
    int col = (nt << 4) + r;
    float x = alpha * c[v];
    if (bias)  x += bias[col];
    if (resid) x += resid[(long long)row * ldc + col];
    if (ACT == 1) x = sspf(x);
    C[(long long)row * ldc + col] = x;
  }
}

// ---------------- fused edge filter: W = ssp(gauss(d)@mw1+b1)@mw2+b2, *cc ----
// mw1/mw2 staged into LDS once per block via the Tensor Data Mover.
__global__ void edge_filter(int nE, const int* __restrict__ sI, const int* __restrict__ dI,
                            const float* __restrict__ posA, const float* __restrict__ vf,
                            const float* __restrict__ mw1, const float* __restrict__ mb1,
                            const float* __restrict__ mw2, const float* __restrict__ mb2,
                            float* __restrict__ Wout)
{
  __shared__ __align__(16) __bf16 hid[8][16][Ff + 8];
  __shared__ __align__(16) float wm1[Gg * Ff];
  __shared__ __align__(16) float wm2[Ff * Ff];
  int wv = threadIdx.x >> 5;
  if (wv == 0) {                       // wave 0 issues the DMA, waits for completion
    tdm_load_f32(mw1, lds_off(wm1), Gg * Ff);
    tdm_load_f32(mw2, lds_off(wm2), Ff * Ff);
    __builtin_amdgcn_s_wait_tensorcnt(0);
  }
  __syncthreads();
  int tile = blockIdx.x * 8 + wv;
  int lane = threadIdx.x & 31, half = lane >> 4, r = lane & 15;
  bool active = (tile << 4) < nE;
  float cc = 0.f;
  if (active) {
    int e = (tile << 4) + r;
    int s = sI[e], t = dI[e];
    float dx = posA[s*3+0]-posA[t*3+0], dy = posA[s*3+1]-posA[t*3+1], dz = posA[s*3+2]-posA[t*3+2];
    float d = sqrtf(dx*dx + dy*dy + dz*dz + 1e-12f);
    cc = ccutf(d);
    if (vf) cc *= vf[e];
    bf16x16 a0, a1;
#pragma unroll
    for (int j = 0; j < 16; ++j) {
      int k = kmapf(j, half), k1 = 32 + k;
      float t0 = d - (float)k * GSTEP, t1v = d - (float)k1 * GSTEP;
      a0[j] = (k  < Gg) ? (__bf16)__expf(GCOEFF * t0 * t0)   : (__bf16)0.f;
      a1[j] = (k1 < Gg) ? (__bf16)__expf(GCOEFF * t1v * t1v) : (__bf16)0.f;
    }
#pragma unroll
    for (int nt = 0; nt < 4; ++nt) {
      int n = (nt << 4) + r;
      floatx8 c = {};
      bf16x16 b0, b1;
#pragma unroll
      for (int j = 0; j < 16; ++j) {
        int k = kmapf(j, half), k1 = 32 + k;
        b0[j] = (k  < Gg) ? (__bf16)wm1[k  * Ff + n] : (__bf16)0.f;
        b1[j] = (k1 < Gg) ? (__bf16)wm1[k1 * Ff + n] : (__bf16)0.f;
      }
      c = __builtin_amdgcn_wmma_f32_16x16x32_bf16(false, a0, false, b0, (short)0, c, false, false);
      c = __builtin_amdgcn_wmma_f32_16x16x32_bf16(false, a1, false, b1, (short)0, c, false, false);
#pragma unroll
      for (int v = 0; v < 8; ++v)
        hid[wv][v + 8 * half][n] = (__bf16)sspf(c[v] + mb1[n]);
    }
  }
  __syncthreads();
  if (active) {
    const bf16x8* rowp = (const bf16x8*)&hid[wv][r][0];   // row is 16B-aligned, stride 144B
#pragma unroll
    for (int nt = 0; nt < 4; ++nt) {
      int n = (nt << 4) + r;
      floatx8 c = {};
#pragma unroll
      for (int k0i = 0; k0i < 2; ++k0i) {
        bf16x8 lo = rowp[k0i * 4 + half];
        bf16x8 hi = rowp[k0i * 4 + 2 + half];
        bf16x16 a = __builtin_shufflevector(lo, hi, 0,1,2,3,4,5,6,7,8,9,10,11,12,13,14,15);
        bf16x16 b;
#pragma unroll
        for (int j = 0; j < 16; ++j) {
          int k = k0i * 32 + kmapf(j, half);
          b[j] = (__bf16)wm2[k * Ff + n];
        }
        c = __builtin_amdgcn_wmma_f32_16x16x32_bf16(false, a, false, b, (short)0, c, false, false);
      }
#pragma unroll
      for (int v = 0; v < 8; ++v) {
        int m = v + 8 * half;
        float ccm = __shfl(cc, m);
        Wout[(long long)((tile << 4) + m) * Ff + n] = (c[v] + mb2[n]) * ccm;
      }
    }
  }
}

// ------- pair score: score[i,j] = relu(hi[i]+hj[j]+gauss(d_ij)@Wd+ab1)·aw2+ab2
__global__ void pair_score_k(const float* __restrict__ pm, const float* __restrict__ hi,
                             const float* __restrict__ hj, const float* __restrict__ Wd,
                             const float* __restrict__ ab1, const float* __restrict__ aw2,
                             const float* __restrict__ ab2p, float* __restrict__ scoreP)
{
  __shared__ __align__(16) float wd[Gg * Hh];
  int wv = threadIdx.x >> 5;
  if (wv == 0) {
    tdm_load_f32(Wd, lds_off(wd), Gg * Hh);
    __builtin_amdgcn_s_wait_tensorcnt(0);
  }
  __syncthreads();
  int tile = blockIdx.x * 8 + wv;
  int lane = threadIdx.x & 31, half = lane >> 4, r = lane & 15;
  int i = tile >> 5;            // Kk/16 = 32 tiles per row
  int jb = (tile & 31) << 4;
  int j0 = jb + r;
  float dx = pm[i*3+0]-pm[j0*3+0], dy = pm[i*3+1]-pm[j0*3+1], dz = pm[i*3+2]-pm[j0*3+2];
  float d = sqrtf(dx*dx + dy*dy + dz*dz + 1e-12f);
  bf16x16 a0, a1;
#pragma unroll
  for (int j = 0; j < 16; ++j) {
    int k = kmapf(j, half), k1 = 32 + k;
    float t0 = d - (float)k * GSTEP, t1v = d - (float)k1 * GSTEP;
    a0[j] = (k  < Gg) ? (__bf16)__expf(GCOEFF * t0 * t0)   : (__bf16)0.f;
    a1[j] = (k1 < Gg) ? (__bf16)__expf(GCOEFF * t1v * t1v) : (__bf16)0.f;
  }
  float s[8];
#pragma unroll
  for (int v = 0; v < 8; ++v) s[v] = 0.f;
#pragma unroll
  for (int nt = 0; nt < 8; ++nt) {
    int n = (nt << 4) + r;
    floatx8 c = {};
    bf16x16 b0, b1;
#pragma unroll
    for (int j = 0; j < 16; ++j) {
      int k = kmapf(j, half), k1 = 32 + k;
      b0[j] = (k  < Gg) ? (__bf16)wd[k  * Hh + n] : (__bf16)0.f;
      b1[j] = (k1 < Gg) ? (__bf16)wd[k1 * Hh + n] : (__bf16)0.f;
    }
    c = __builtin_amdgcn_wmma_f32_16x16x32_bf16(false, a0, false, b0, (short)0, c, false, false);
    c = __builtin_amdgcn_wmma_f32_16x16x32_bf16(false, a1, false, b1, (short)0, c, false, false);
    float hin = hi[(long long)i * Hh + n];
    float b1n = ab1[n], w2n = aw2[n];
#pragma unroll
    for (int v = 0; v < 8; ++v) {
      int m = v + 8 * half;
      float p = c[v] + hin + hj[(long long)(jb + m) * Hh + n] + b1n;
      s[v] += fmaxf(p, 0.f) * w2n;
    }
  }
#pragma unroll
  for (int off = 1; off < 16; off <<= 1)
#pragma unroll
    for (int v = 0; v < 8; ++v) s[v] += __shfl_xor(s[v], off);
  if (r == 0) {
    float b2 = ab2p[0];
#pragma unroll
    for (int v = 0; v < 8; ++v)
      scoreP[(long long)i * Kk + jb + v + 8 * half] = s[v] + b2;
  }
}

// ---- pair Wv + agg_d: Wv=filt(gauss*decay)*(ccut*Av); agg_d[j]+=xm[i]*Wv ----
__global__ void pair_aggd_k(const float* __restrict__ pm, const float* __restrict__ decay,
                            const float* __restrict__ Av, const float* __restrict__ mw1,
                            const float* __restrict__ mb1, const float* __restrict__ mw2,
                            const float* __restrict__ mb2, const float* __restrict__ xm,
                            float* __restrict__ agg_d)
{
  __shared__ __align__(16) __bf16 hid[8][16][Ff + 8];
  __shared__ __align__(16) float wm1[Gg * Ff];
  __shared__ __align__(16) float wm2[Ff * Ff];
  int wv = threadIdx.x >> 5;
  if (wv == 0) {
    tdm_load_f32(mw1, lds_off(wm1), Gg * Ff);
    tdm_load_f32(mw2, lds_off(wm2), Ff * Ff);
    __builtin_amdgcn_s_wait_tensorcnt(0);
  }
  __syncthreads();
  int tile = blockIdx.x * 8 + wv;
  int lane = threadIdx.x & 31, half = lane >> 4, r = lane & 15;
  int i = tile >> 5;
  int jb = (tile & 31) << 4;
  int j0 = jb + r;
  float av = Av[(long long)i * Kk + j0];
  bool any = (__ballot(av > 0.f) != 0ull);
  float cc = 0.f;
  if (any) {
    float dx = pm[i*3+0]-pm[j0*3+0], dy = pm[i*3+1]-pm[j0*3+1], dz = pm[i*3+2]-pm[j0*3+2];
    float d = sqrtf(dx*dx + dy*dy + dz*dz + 1e-12f);
    cc = ccutf(d) * av;
    float dec = decay[(long long)i * Kk + j0];
    bf16x16 a0, a1;
#pragma unroll
    for (int j = 0; j < 16; ++j) {
      int k = kmapf(j, half), k1 = 32 + k;
      float t0 = d - (float)k * GSTEP, t1v = d - (float)k1 * GSTEP;
      a0[j] = (k  < Gg) ? (__bf16)(__expf(GCOEFF * t0 * t0)  * dec) : (__bf16)0.f;
      a1[j] = (k1 < Gg) ? (__bf16)(__expf(GCOEFF * t1v * t1v) * dec) : (__bf16)0.f;
    }
#pragma unroll
    for (int nt = 0; nt < 4; ++nt) {
      int n = (nt << 4) + r;
      floatx8 c = {};
      bf16x16 b0, b1;
#pragma unroll
      for (int j = 0; j < 16; ++j) {
        int k = kmapf(j, half), k1 = 32 + k;
        b0[j] = (k  < Gg) ? (__bf16)wm1[k  * Ff + n] : (__bf16)0.f;
        b1[j] = (k1 < Gg) ? (__bf16)wm1[k1 * Ff + n] : (__bf16)0.f;
      }
      c = __builtin_amdgcn_wmma_f32_16x16x32_bf16(false, a0, false, b0, (short)0, c, false, false);
      c = __builtin_amdgcn_wmma_f32_16x16x32_bf16(false, a1, false, b1, (short)0, c, false, false);
#pragma unroll
      for (int v = 0; v < 8; ++v)
        hid[wv][v + 8 * half][n] = (__bf16)sspf(c[v] + mb1[n]);
    }
  }
  __syncthreads();
  if (any) {
    const bf16x8* rowp = (const bf16x8*)&hid[wv][r][0];
#pragma unroll
    for (int nt = 0; nt < 4; ++nt) {
      int n = (nt << 4) + r;
      floatx8 c = {};
#pragma unroll
      for (int k0i = 0; k0i < 2; ++k0i) {
        bf16x8 lo = rowp[k0i * 4 + half];
        bf16x8 hi = rowp[k0i * 4 + 2 + half];
        bf16x16 a = __builtin_shufflevector(lo, hi, 0,1,2,3,4,5,6,7,8,9,10,11,12,13,14,15);
        bf16x16 b;
#pragma unroll
        for (int j = 0; j < 16; ++j) {
          int k = k0i * 32 + kmapf(j, half);
          b[j] = (__bf16)wm2[k * Ff + n];
        }
        c = __builtin_amdgcn_wmma_f32_16x16x32_bf16(false, a, false, b, (short)0, c, false, false);
      }
      float xin = xm[(long long)i * Ff + n];
#pragma unroll
      for (int v = 0; v < 8; ++v) {
        int m = v + 8 * half, jj = jb + m;
        float ccm = __shfl(cc, m);
        float w = (c[v] + mb2[n]) * ccm;
        if (w != 0.f) atomicAdd(&agg_d[(long long)jj * Ff + n], xin * w);
      }
    }
  }
}

// ---------------- small supporting kernels ----------------
__global__ void embed_k(const int* __restrict__ atoms, const float* __restrict__ emb,
                        float* __restrict__ h) {
  int t = blockIdx.x * blockDim.x + threadIdx.x;
  if (t >= Nn * Hh) return;
  int n = t >> 7, j = t & 127;
  h[t] = emb[atoms[n] * Hh + j];
}
__global__ void fill_k(float* __restrict__ p, long long n, float v) {
  long long t = (long long)blockIdx.x * blockDim.x + threadIdx.x;
  if (t < n) p[t] = v;
}
__global__ void add_k(const float* a, const float* b, float* c, int n) {
  int t = blockIdx.x * blockDim.x + threadIdx.x;
  if (t < n) c[t] = a[t] + b[t];
}
__global__ void scatter_agg_k(int nE, const int* __restrict__ sI, const int* __restrict__ dI,
                              const float* __restrict__ x, const float* __restrict__ W,
                              float* __restrict__ agg) {
  long long t = (long long)blockIdx.x * blockDim.x + threadIdx.x;
  if (t >= (long long)nE * Ff) return;
  int e = (int)(t >> 6), f = (int)(t & 63);
  float w = W[t];
  if (w != 0.f) atomicAdd(&agg[(long long)dI[e] * Ff + f], x[(long long)sI[e] * Ff + f] * w);
}
__global__ void ms_score_k(const float* __restrict__ h_local, const float* __restrict__ w1,
                           const float* __restrict__ b1, const float* __restrict__ w2,
                           const float* __restrict__ b2p, float* __restrict__ scores) {
  int n = blockIdx.x * blockDim.x + threadIdx.x;
  if (n >= Nn) return;
  float hs[Ss];
#pragma unroll
  for (int s = 0; s < Ss; ++s) hs[s] = h_local[(long long)n * Hh + s];
  float acc = b2p[0];
  for (int mh = 0; mh < MHh; ++mh) {
    float z = b1[mh];
#pragma unroll
    for (int s = 0; s < Ss; ++s) z += hs[s] * w1[s * MHh + mh];
    acc += fmaxf(z, 0.f) * w2[mh];
  }
  scores[n] = acc;
}
__global__ void topk_k(const float* __restrict__ scores, float* __restrict__ mval,
                       int* __restrict__ midx, int* __restrict__ inv, int* __restrict__ ism) {
  __shared__ float key[Nn];
  __shared__ int   idx[Nn];
  __shared__ int   mi[Kk];
  int t = threadIdx.x;
  for (int n = t; n < Nn; n += 1024) { key[n] = -scores[n]; idx[n] = n; }
  __syncthreads();
  for (int kk2 = 2; kk2 <= Nn; kk2 <<= 1) {
    for (int jj = kk2 >> 1; jj > 0; jj >>= 1) {
      for (int n = t; n < Nn; n += 1024) {
        int p = n ^ jj;
        if (p > n) {
          bool up = ((n & kk2) == 0);
          float kn = key[n], kp = key[p];
          if ((kn > kp) == up) {
            key[n] = kp; key[p] = kn;
            int tm = idx[n]; idx[n] = idx[p]; idx[p] = tm;
          }
        }
      }
      __syncthreads();
    }
  }
  float kth = -key[Kk - 1];   // K-th largest score
  if (t < Kk) mi[t] = idx[t];
  __syncthreads();
  for (int kk2 = 2; kk2 <= Kk; kk2 <<= 1) {
    for (int jj = kk2 >> 1; jj > 0; jj >>= 1) {
      if (t < Kk) {
        int p = t ^ jj;
        if (p > t) {
          bool up = ((t & kk2) == 0);
          int an = mi[t], ap = mi[p];
          if ((an > ap) == up) { mi[t] = ap; mi[p] = an; }
        }
      }
      __syncthreads();
    }
  }
  for (int n = t; n < Nn; n += 1024) {
    ism[n] = 0; inv[n] = 0;
    mval[n] = sigmf(scores[n] - kth + 1e-6f);
  }
  __syncthreads();
  if (t < Kk) { midx[t] = mi[t]; ism[mi[t]] = 1; inv[mi[t]] = t; }
}
__global__ void edge_mask_k(const int* src, const int* dst, const int* ism, const int* inv,
                            int* srcm, int* dstm, float* vf, float* adjc) {
  int e = blockIdx.x * blockDim.x + threadIdx.x;
  if (e >= Ee) return;
  int v = ism[src[e]] & ism[dst[e]];
  int sm = v ? inv[src[e]] : 0;
  int dm = v ? inv[dst[e]] : 0;
  srcm[e] = sm; dstm[e] = dm; vf[e] = (float)v;
  if (v) atomicAdd(&adjc[(long long)sm * Kk + dm], 1.0f);
}
__global__ void gather_hm_k(const float* h_local, const float* pos, const int* midx,
                            float* h_m, float* pos_m) {
  int t = blockIdx.x * blockDim.x + threadIdx.x;
  if (t >= Kk * Hh) return;
  int k = t >> 7, j = t & 127;
  int n = midx[k];
  h_m[t] = h_local[(long long)n * Hh + j];
  if (j < 3) pos_m[k * 3 + j] = pos[n * 3 + j];
}
__global__ void softmax_row_k(const float* __restrict__ in, const float* __restrict__ mask,
                              float* __restrict__ outp, int len) {
  __shared__ float red[256];
  int i = blockIdx.x, t = threadIdx.x;
  const float* rowi = in + (long long)i * len;
  const float* mrow = mask ? mask + (long long)i * len : nullptr;
  float mx = -1e30f;
  for (int j = t; j < len; j += 256) {
    float v = rowi[j];
    if (mrow) v = (mrow[j] > 0.f) ? v : -1e9f;
    mx = fmaxf(mx, v);
  }
  red[t] = mx; __syncthreads();
  for (int s2 = 128; s2 > 0; s2 >>= 1) { if (t < s2) red[t] = fmaxf(red[t], red[t + s2]); __syncthreads(); }
  mx = red[0]; __syncthreads();
  float sum = 0.f;
  for (int j = t; j < len; j += 256) {
    float v = rowi[j];
    if (mrow) v = (mrow[j] > 0.f) ? v : -1e9f;
    sum += __expf(v - mx);
  }
  red[t] = sum; __syncthreads();
  for (int s2 = 128; s2 > 0; s2 >>= 1) { if (t < s2) red[t] += red[t + s2]; __syncthreads(); }
  sum = red[0];
  float isum = 1.f / sum;
  for (int j = t; j < len; j += 256) {
    float v = rowi[j];
    float mk = 1.f;
    if (mrow) { mk = mrow[j]; v = (mk > 0.f) ? v : -1e9f; }
    outp[(long long)i * len + j] = __expf(v - mx) * isum * mk;
  }
}
__global__ void compute_av_k(const float* attn, const float* adjc, float* Av) {
  long long t = (long long)blockIdx.x * blockDim.x + threadIdx.x;
  if (t >= (long long)Kk * Kk) return;
  int i = (int)(t >> 9), j = (int)(t & 511);
  Av[t] = (attn[t] > LAMBDA_A && adjc[t] <= 0.f && i != j) ? 1.f : 0.f;
}
__global__ void combine_h_k(const float* h_local, const float* h_hier, const float* mval,
                            const int* ism, const int* inv, float* h_out) {
  int t = blockIdx.x * blockDim.x + threadIdx.x;
  if (t >= Nn * Hh) return;
  int n = t >> 7, j = t & 127;
  float mm = mval[n];
  float v = (1.f - mm) * h_local[t];
  if (ism[n]) v += mm * h_hier[(long long)inv[n] * Hh + j];
  h_out[t] = v;
}
__global__ void pool_k(const float* h, const int* batch, float* pooled) {
  int t = blockIdx.x * blockDim.x + threadIdx.x;
  if (t >= Nn * Hh) return;
  int n = t >> 7, j = t & 127;
  atomicAdd(&pooled[(long long)batch[n] * Hh + j], h[t]);
}
__global__ void pred_head_k(const float* pooled, const float* w1, const float* b1,
                            const float* w2, const float* b2, float* outp) {
  __shared__ float red[64];
  int b = blockIdx.x, j = threadIdx.x;
  float z = b1[j];
  for (int hh = 0; hh < Hh; ++hh) z += pooled[b * Hh + hh] * w1[hh * 64 + j];
  float sil = z * sigmf(z);
  red[j] = sil * w2[j];
  __syncthreads();
  for (int s2 = 32; s2 > 0; s2 >>= 1) { if (j < s2) red[j] += red[j + s2]; __syncthreads(); }
  if (j == 0) outp[b] = red[0] + b2[0];
}

// ---------------- host orchestration ----------------
extern "C" void kernel_launch(void* const* d_in, const int* in_sizes, int n_in,
                              void* d_out, int out_size, void* d_ws, size_t ws_size,
                              hipStream_t stream)
{
  const int*   atoms = (const int*)d_in[0];
  const float* pos   = (const float*)d_in[1];
  const int*   src   = (const int*)d_in[2];
  const int*   dst   = (const int*)d_in[3];
  const int*   batch = (const int*)d_in[4];
  const float* emb   = (const float*)d_in[5];
  const float* ib_mlp_w1 = (const float*)d_in[6];
  const float* ib_mlp_b1 = (const float*)d_in[7];
  const float* ib_mlp_w2 = (const float*)d_in[8];
  const float* ib_mlp_b2 = (const float*)d_in[9];
  const float* cf_lin1_w = (const float*)d_in[10];
  const float* cf_lin2_w = (const float*)d_in[11];
  const float* cf_lin2_b = (const float*)d_in[12];
  const float* ib_lin_w  = (const float*)d_in[13];
  const float* ib_lin_b  = (const float*)d_in[14];
  const float* attn_w1   = (const float*)d_in[15];
  const float* attn_b1   = (const float*)d_in[16];
  const float* attn_w2   = (const float*)d_in[17];
  const float* attn_b2   = (const float*)d_in[18];
  const float* ms_w1     = (const float*)d_in[19];
  const float* ms_b1     = (const float*)d_in[20];
  const float* ms_w2     = (const float*)d_in[21];
  const float* ms_b2     = (const float*)d_in[22];
  const float* vg_wq     = (const float*)d_in[23];
  const float* vg_wk     = (const float*)d_in[24];
  const float* pred_w1   = (const float*)d_in[25];
  const float* pred_b1   = (const float*)d_in[26];
  const float* pred_w2   = (const float*)d_in[27];
  const float* pred_b2   = (const float*)d_in[28];
  float* outv = (float*)d_out;
  (void)in_sizes; (void)n_in; (void)out_size; (void)ws_size;

  char* wsb = (char*)d_ws;
  size_t off = 0;
  auto alloc = [&](size_t bytes) -> void* {
    void* p = wsb + off;
    off += (bytes + 255) & ~(size_t)255;
    return p;
  };
  float* h0      = (float*)alloc((size_t)Nn*Hh*4);
  float* h1      = (float*)alloc((size_t)Nn*Hh*4);
  float* h_local = (float*)alloc((size_t)Nn*Hh*4);
  float* t1      = (float*)alloc((size_t)Nn*Hh*4);
  float* xbuf    = (float*)alloc((size_t)Nn*Ff*4);
  float* Wedge   = (float*)alloc((size_t)Ee*Ff*4);
  float* agg     = (float*)alloc((size_t)Nn*Ff*4);
  float* scores  = (float*)alloc((size_t)Nn*4);
  float* mval    = (float*)alloc((size_t)Nn*4);
  int*   midx    = (int*)alloc((size_t)Kk*4);
  int*   invb    = (int*)alloc((size_t)Nn*4);
  int*   ism     = (int*)alloc((size_t)Nn*4);
  int*   srcm    = (int*)alloc((size_t)Ee*4);
  int*   dstm    = (int*)alloc((size_t)Ee*4);
  float* vfb     = (float*)alloc((size_t)Ee*4);
  float* adjc    = (float*)alloc((size_t)Kk*Kk*4);
  float* h_m     = (float*)alloc((size_t)Kk*Hh*4);
  float* pos_m   = (float*)alloc((size_t)Kk*3*4);
  float* xm      = (float*)alloc((size_t)Kk*Ff*4);
  float* qb      = (float*)alloc((size_t)Kk*Ss*4);
  float* kb      = (float*)alloc((size_t)Kk*Ss*4);
  float* attnB   = (float*)alloc((size_t)Kk*Kk*4);
  float* AvB     = (float*)alloc((size_t)Kk*Kk*4);
  float* scoreP  = (float*)alloc((size_t)Kk*Kk*4);
  float* decayB  = (float*)alloc((size_t)Kk*Kk*4);
  float* hiB     = (float*)alloc((size_t)Kk*Hh*4);
  float* hjB     = (float*)alloc((size_t)Kk*Hh*4);
  float* agg_s   = (float*)alloc((size_t)Kk*Ff*4);
  float* agg_d   = (float*)alloc((size_t)Kk*Ff*4);
  float* aggm    = (float*)alloc((size_t)Kk*Ff*4);
  float* t1m     = (float*)alloc((size_t)Kk*Hh*4);
  float* h_hier  = (float*)alloc((size_t)Kk*Hh*4);
  float* pooled  = (float*)alloc((size_t)Bb*Hh*4);

  embed_k<<<(Nn*Hh)/256, 256, 0, stream>>>(atoms, emb, h0);

  float* hcur = h0; float* hnext = h1;
  for (int l = 0; l < 2; ++l) {
    const float* mw1  = ib_mlp_w1 + (size_t)l*Gg*Ff;
    const float* mb1  = ib_mlp_b1 + (size_t)l*Ff;
    const float* mw2  = ib_mlp_w2 + (size_t)l*Ff*Ff;
    const float* mb2  = ib_mlp_b2 + (size_t)l*Ff;
    const float* cf1  = cf_lin1_w + (size_t)l*Hh*Ff;
    const float* cf2w = cf_lin2_w + (size_t)l*Ff*Hh;
    const float* cf2b = cf_lin2_b + (size_t)l*Hh;
    const float* lwp  = ib_lin_w  + (size_t)l*Hh*Hh;
    const float* lbp  = ib_lin_b  + (size_t)l*Hh;
    const float* aw1  = attn_w1   + (size_t)l*(2*Hh+Gg)*Hh;
    const float* ab1  = attn_b1   + (size_t)l*Hh;
    const float* aw2  = attn_w2   + (size_t)l*Hh;
    const float* ab2p = attn_b2   + l;
    const float* msw1 = ms_w1     + (size_t)l*Ss*MHh;
    const float* msb1 = ms_b1     + (size_t)l*MHh;
    const float* msw2 = ms_w2     + (size_t)l*MHh;
    const float* msb2p= ms_b2     + l;
    const float* wq   = vg_wq     + (size_t)l*Ss*Ss;
    const float* wk   = vg_wk     + (size_t)l*Ss*Ss;

    // W = filt(gauss(d)) * ccut(d)  (global edges)
    edge_filter<<<Ee/16/8, 256, 0, stream>>>(Ee, src, dst, pos, nullptr, mw1, mb1, mw2, mb2, Wedge);
    // x = h @ cf1
    gemm_ep_t<0,0><<<(128*4+7)/8, 256, 0, stream>>>(hcur,Hh, cf1,Ff, nullptr,nullptr, xbuf,Ff, Nn,Hh,Ff, 1.f);
    fill_k<<<(Nn*Ff+255)/256, 256, 0, stream>>>(agg, (long long)Nn*Ff, 0.f);
    scatter_agg_k<<<(Ee*Ff)/256, 256, 0, stream>>>(Ee, src, dst, xbuf, Wedge, agg);
    // h_local = h + ssp(agg@cf2w+cf2b)@lw + lb
    gemm_ep_t<0,1><<<(128*8+7)/8, 256, 0, stream>>>(agg,Ff, cf2w,Hh, cf2b,nullptr, t1,Hh, Nn,Ff,Hh, 1.f);
    gemm_ep_t<0,0><<<(128*8+7)/8, 256, 0, stream>>>(t1,Hh, lwp,Hh, lbp,hcur, h_local,Hh, Nn,Hh,Hh, 1.f);
    // mincut scores + top-K
    ms_score_k<<<Nn/256, 256, 0, stream>>>(h_local, msw1, msb1, msw2, msb2p, scores);
    topk_k<<<1, 1024, 0, stream>>>(scores, mval, midx, invb, ism);
    fill_k<<<((long long)Kk*Kk+255)/256, 256, 0, stream>>>(adjc, (long long)Kk*Kk, 0.f);
    edge_mask_k<<<Ee/256, 256, 0, stream>>>(src, dst, ism, invb, srcm, dstm, vfb, adjc);
    gather_hm_k<<<(Kk*Hh)/256, 256, 0, stream>>>(h_local, pos, midx, h_m, pos_m);
    // xm, q, k, attention
    gemm_ep_t<0,0><<<(32*4+7)/8, 256, 0, stream>>>(h_m,Hh, cf1,Ff, nullptr,nullptr, xm,Ff, Kk,Hh,Ff, 1.f);
    gemm_ep_t<0,0><<<(32*1+7)/8, 256, 0, stream>>>(h_m,Hh, wq,Ss, nullptr,nullptr, qb,Ss, Kk,Ss,Ss, 1.f);
    gemm_ep_t<0,0><<<(32*1+7)/8, 256, 0, stream>>>(h_m,Hh, wk,Ss, nullptr,nullptr, kb,Ss, Kk,Ss,Ss, 1.f);
    gemm_ep_t<1,0><<<(32*32+7)/8, 256, 0, stream>>>(qb,Ss, kb,Ss, nullptr,nullptr, attnB,Kk, Kk,Ss,Kk, 0.25f);
    softmax_row_k<<<Kk, 256, 0, stream>>>(attnB, nullptr, attnB, Kk);
    compute_av_k<<<((long long)Kk*Kk)/256, 256, 0, stream>>>(attnB, adjc, AvB);
    // mini-graph edge filter + agg_s
    edge_filter<<<Ee/16/8, 256, 0, stream>>>(Ee, srcm, dstm, pos_m, vfb, mw1, mb1, mw2, mb2, Wedge);
    fill_k<<<(Kk*Ff+255)/256, 256, 0, stream>>>(agg_s, (long long)Kk*Ff, 0.f);
    scatter_agg_k<<<(Ee*Ff)/256, 256, 0, stream>>>(Ee, srcm, dstm, xm, Wedge, agg_s);
    // pairwise attention-decayed aggregation
    gemm_ep_t<0,0><<<(32*8+7)/8, 256, 0, stream>>>(h_m,Hh, aw1,Hh, nullptr,nullptr, hiB,Hh, Kk,Hh,Hh, 1.f);
    gemm_ep_t<0,0><<<(32*8+7)/8, 256, 0, stream>>>(h_m,Hh, aw1+(size_t)Hh*Hh,Hh, nullptr,nullptr, hjB,Hh, Kk,Hh,Hh, 1.f);
    pair_score_k<<<(Kk*Kk/16)/8, 256, 0, stream>>>(pos_m, hiB, hjB, aw1+(size_t)2*Hh*Hh, ab1, aw2, ab2p, scoreP);
    softmax_row_k<<<Kk, 256, 0, stream>>>(scoreP, AvB, decayB, Kk);
    fill_k<<<(Kk*Ff+255)/256, 256, 0, stream>>>(agg_d, (long long)Kk*Ff, 0.f);
    pair_aggd_k<<<(Kk*Kk/16)/8, 256, 0, stream>>>(pos_m, decayB, AvB, mw1, mb1, mw2, mb2, xm, agg_d);
    // h_hier and recombination
    add_k<<<(Kk*Ff+255)/256, 256, 0, stream>>>(agg_s, agg_d, aggm, Kk*Ff);
    gemm_ep_t<0,1><<<(32*8+7)/8, 256, 0, stream>>>(aggm,Ff, cf2w,Hh, cf2b,nullptr, t1m,Hh, Kk,Ff,Hh, 1.f);
    gemm_ep_t<0,0><<<(32*8+7)/8, 256, 0, stream>>>(t1m,Hh, lwp,Hh, lbp,h_m, h_hier,Hh, Kk,Hh,Hh, 1.f);
    combine_h_k<<<(Nn*Hh)/256, 256, 0, stream>>>(h_local, h_hier, mval, ism, invb, hnext);
    float* tmp = hcur; hcur = hnext; hnext = tmp;
  }
  fill_k<<<(Bb*Hh+255)/256, 256, 0, stream>>>(pooled, (long long)Bb*Hh, 0.f);
  pool_k<<<(Nn*Hh)/256, 256, 0, stream>>>(hcur, batch, pooled);
  pred_head_k<<<Bb, 64, 0, stream>>>(pooled, pred_w1, pred_b1, pred_w2, pred_b2, outv);
}
